// NeuralODEModel_38001870635348
// MI455X (gfx1250) — compile-verified, matrix operand order checked by reading, not run on hardware
//
#include <hip/hip_runtime.h>
#include <stdint.h>
#include <stddef.h>

// ---------------------------------------------------------------------------
// Types for CDNA5 WMMA (wave32): v_wmma_f32_16x16x32_bf16
// ---------------------------------------------------------------------------
typedef __attribute__((ext_vector_type(16))) __bf16 v16bf;
typedef __attribute__((ext_vector_type(8)))  __bf16 v8bf;
typedef __attribute__((ext_vector_type(8)))  float  v8f;
typedef __attribute__((ext_vector_type(4)))  int    v4i;

union Frag  { uint4 q[2]; v16bf v; };   // 32B = one 16x32 bf16 fragment per lane
union Pack8 { v8bf h; uint4 q; };       // 8 bf16 <-> 16 bytes

#if __has_builtin(__builtin_amdgcn_global_load_async_to_lds_b128) && \
    __has_builtin(__builtin_amdgcn_s_wait_asynccnt)
#define USE_ASYNC_LDS 1
#else
#define USE_ASYNC_LDS 0
#endif

// copy 8 bf16 (16B) global -> LDS; async (ASYNCcnt) when available
__device__ __forceinline__ void copy_row8(const unsigned short* __restrict__ src,
                                          unsigned short* dst) {
#if USE_ASYNC_LDS
    __builtin_amdgcn_global_load_async_to_lds_b128(
        (__attribute__((address_space(1))) v4i*)src,
        (__attribute__((address_space(3))) v4i*)dst,
        0, 0);
#else
    *(uint4*)dst = *(const uint4*)src;
#endif
}

// 8 elements of bf16(y + c*k), hardware cvt (v_cvt_pk_bf16_f32)
__device__ __forceinline__ uint4 fuse8(const float* __restrict__ yp,
                                       const float* __restrict__ kp, float c) {
    float4 y0 = *(const float4*)yp;
    float4 y1 = *(const float4*)(yp + 4);
    float4 k0 = *(const float4*)kp;
    float4 k1 = *(const float4*)(kp + 4);
    v8f u;
    u[0] = y0.x + c * k0.x; u[1] = y0.y + c * k0.y;
    u[2] = y0.z + c * k0.z; u[3] = y0.w + c * k0.w;
    u[4] = y1.x + c * k1.x; u[5] = y1.y + c * k1.y;
    u[6] = y1.z + c * k1.z; u[7] = y1.w + c * k1.w;
    Pack8 p;
    p.h = __builtin_convertvector(u, v8bf);
    return p.q;
}

__device__ __forceinline__ unsigned short bf16s(float a) {
    union { __bf16 h; unsigned short s; } u;
    u.h = (__bf16)a;
    return u.s;
}

// ---------------------------------------------------------------------------
// GEMM: C(256 x N) = A(256 x 512) * B(N x 512)^T + bias, K = 512.
// Block: 128 threads = 4 waves (2x2), 64x64 tile; wave tile 32x32 (2x2 WMMA).
// K staged in 4 quarters of 128 into LDS (A and B), +8 bf16 row pad for
// conflict-free ds_load_b128 fragment reads.
// AFUSED: A tile built on the fly as bf16(y + cmul*dt*k) from f32.
// MODE 0: out = bf16(tanh(C))
// MODE 1: out = C (f32 k_j); Yout = (first? Yin : Yout) + wmul*dt*C
// MODE 2: out = C (f32)
// ---------------------------------------------------------------------------
template <int AFUSED, int MODE>
__global__ __launch_bounds__(128) void gemm_kernel(
    const unsigned short* __restrict__ Abf,
    const float* __restrict__ Yf, const float* __restrict__ Kf, float cmul,
    const unsigned short* __restrict__ Bm,
    const float* __restrict__ bias, int N,
    unsigned short* __restrict__ outB,
    float* __restrict__ outF,
    const float* __restrict__ Yin, float* __restrict__ Yout,
    float wmul, int first,
    const float* __restrict__ x, int step)
{
    constexpr int PITCH = 136;   // 128 + 8 pad (bf16 elems); 272B rows, 16B aligned
    __shared__ __align__(16) unsigned short As[64 * PITCH];
    __shared__ __align__(16) unsigned short Bs[64 * PITCH];

    const int tid  = threadIdx.x;
    const int lane = tid & 31;
    const int w    = tid >> 5;
    const int wm   = w & 1;
    const int wn   = w >> 1;
    const int lr   = lane & 15;
    const int kb   = (lane >> 4) << 3;     // fragment K interleave base

    const int rowBlk = blockIdx.y * 64;
    const int colBlk = blockIdx.x * 64;

    const float dt = x[(size_t)(step + 1) * 513] - x[(size_t)step * 513];
    const float c  = AFUSED ? (cmul * dt) : 0.0f;

    v8f acc00 = {}, acc01 = {}, acc10 = {}, acc11 = {};

    // staging coords: thread t handles rows (t>>4)+8*it at 8-elem chunk (t&15)*8
    const int sr = tid >> 4;
    const int sq = (tid & 15) * 8;

    for (int qb = 0; qb < 512; qb += 128) {
        // ---- stage B (weights, pure copy; async overlaps with A conversion) ----
#pragma unroll
        for (int it = 0; it < 8; ++it) {
            const int r = sr + it * 8;
            copy_row8(Bm + (size_t)(colBlk + r) * 512 + qb + sq, &Bs[r * PITCH + sq]);
        }
        // ---- stage A ----
        if (AFUSED) {
#pragma unroll
            for (int it = 0; it < 8; ++it) {
                const int r = sr + it * 8;
                const size_t g = (size_t)(rowBlk + r) * 512 + qb + sq;
                *(uint4*)&As[r * PITCH + sq] = fuse8(Yf + g, Kf + g, c);
            }
        } else {
#pragma unroll
            for (int it = 0; it < 8; ++it) {
                const int r = sr + it * 8;
                copy_row8(Abf + (size_t)(rowBlk + r) * 512 + qb + sq, &As[r * PITCH + sq]);
            }
        }
#if USE_ASYNC_LDS
        __builtin_amdgcn_s_wait_asynccnt(0);
#endif
        __syncthreads();

        // ---- compute: 4 K-chunks of 32, pure ds_load_b128 + v_wmma ----
#pragma unroll
        for (int kq = 0; kq < 128; kq += 32) {
            const unsigned short* a0 = &As[(wm * 32 + lr) * PITCH + kq + kb];
            const unsigned short* b0 = &Bs[(wn * 32 + lr) * PITCH + kq + kb];
            Frag fa0, fa1, fb0, fb1;
            fa0.q[0] = *(const uint4*)(a0);
            fa0.q[1] = *(const uint4*)(a0 + 16);
            fa1.q[0] = *(const uint4*)(a0 + 16 * PITCH);
            fa1.q[1] = *(const uint4*)(a0 + 16 * PITCH + 16);
            fb0.q[0] = *(const uint4*)(b0);
            fb0.q[1] = *(const uint4*)(b0 + 16);
            fb1.q[0] = *(const uint4*)(b0 + 16 * PITCH);
            fb1.q[1] = *(const uint4*)(b0 + 16 * PITCH + 16);

            acc00 = __builtin_amdgcn_wmma_f32_16x16x32_bf16(false, fa0.v, false, fb0.v,
                                                            (short)0, acc00, false, false);
            acc01 = __builtin_amdgcn_wmma_f32_16x16x32_bf16(false, fa0.v, false, fb1.v,
                                                            (short)0, acc01, false, false);
            acc10 = __builtin_amdgcn_wmma_f32_16x16x32_bf16(false, fa1.v, false, fb0.v,
                                                            (short)0, acc10, false, false);
            acc11 = __builtin_amdgcn_wmma_f32_16x16x32_bf16(false, fa1.v, false, fb1.v,
                                                            (short)0, acc11, false, false);
        }
        __syncthreads();
    }

    // ---- epilogue. f32 C/D: lane<16 -> N=lane, M=j; lane>=16 -> N=lane-16, M=8+j
    const float dtw = wmul * dt;
    const int rlo = (lane >> 4) << 3;
    v8f acc[2][2] = {{acc00, acc01}, {acc10, acc11}};
#pragma unroll
    for (int ms = 0; ms < 2; ++ms) {
#pragma unroll
        for (int ns = 0; ns < 2; ++ns) {
            const int col  = colBlk + wn * 32 + ns * 16 + lr;
            const int row0 = rowBlk + wm * 32 + ms * 16 + rlo;
            const float bv = bias[col];
#pragma unroll
            for (int j = 0; j < 8; ++j) {
                const float v = acc[ms][ns][j] + bv;
                const size_t idx = (size_t)(row0 + j) * (size_t)N + (size_t)col;
                if (MODE == 0) {
                    outB[idx] = bf16s(tanhf(v));
                } else if (MODE == 1) {
                    outF[idx] = v;                                    // k_j
                    const float base = first ? Yin[idx] : Yout[idx];  // RK4 accumulation
                    Yout[idx] = base + dtw * v;
                } else {
                    outF[idx] = v;
                }
            }
        }
    }
}

// ---------------------------------------------------------------------------
// Small helper kernels
// ---------------------------------------------------------------------------
__global__ void cvt_bf16_kernel(const float* __restrict__ src,
                                unsigned short* __restrict__ dst, int n) {
    int i = blockIdx.x * blockDim.x + threadIdx.x;
    if (i < n) dst[i] = bf16s(src[i]);
}

// y0[b,d] = x[b, S-1, 1+d];  x is (256, 256, 513)
__global__ void init_y_kernel(const float* __restrict__ x, float* __restrict__ y) {
    int i = blockIdx.x * blockDim.x + threadIdx.x;   // 256*512 threads
    int b = i >> 9;
    int d = i & 511;
    y[i] = x[(size_t)b * (256 * 513) + (size_t)255 * 513 + 1 + d];
}

// ---------------------------------------------------------------------------
// Host orchestration: 255 RK4 steps x 4 evals x 2 GEMMs, graph-capturable.
// ---------------------------------------------------------------------------
extern "C" void kernel_launch(void* const* d_in, const int* in_sizes, int n_in,
                              void* d_out, int out_size, void* d_ws, size_t ws_size,
                              hipStream_t stream) {
    const float* x    = (const float*)d_in[0];
    const float* W1   = (const float*)d_in[1];
    const float* b1   = (const float*)d_in[2];
    const float* W2   = (const float*)d_in[3];
    const float* b2   = (const float*)d_in[4];
    const float* Wout = (const float*)d_in[5];
    const float* bout = (const float*)d_in[6];
    float* out = (float*)d_out;

    char* ws = (char*)d_ws;
    const size_t YB = (size_t)256 * 512 * sizeof(float);        // 512 KB
    float* Y0 = (float*)(ws);
    float* Y1 = (float*)(ws + YB);
    float* Kb = (float*)(ws + 2 * YB);
    unsigned short* Hbf = (unsigned short*)(ws + 3 * YB);                    // 256 KB
    unsigned short* W1b = (unsigned short*)(ws + 3 * YB + 256 * 1024);      // 512 KB
    unsigned short* W2b = (unsigned short*)(ws + 3 * YB + 768 * 1024);      // 512 KB
    unsigned short* Wob = (unsigned short*)(ws + 3 * YB + 1280 * 1024);     // 128 KB

    // One-time weight conversion + state init
    cvt_bf16_kernel<<<1024, 256, 0, stream>>>(W1, W1b, 512 * 512);
    cvt_bf16_kernel<<<1024, 256, 0, stream>>>(W2, W2b, 512 * 512);
    cvt_bf16_kernel<<<256, 256, 0, stream>>>(Wout, Wob, 128 * 512);
    init_y_kernel<<<512, 256, 0, stream>>>(x, Y0);

    const dim3 grid_main(8, 4);   // N=512: 8 col blocks; M=256: 4 row blocks
    const dim3 grid_out(2, 4);    // N=128
    const dim3 blk(128);

    const float Cc[4] = {0.0f, 0.5f, 0.5f, 1.0f};                        // u = y + c*dt*k
    const float Wg[4] = {1.0f / 6.0f, 1.0f / 3.0f, 1.0f / 3.0f, 1.0f / 6.0f};

    for (int s = 0; s < 255; ++s) {
        float* Yin  = (s & 1) ? Y1 : Y0;
        float* Yout = (s & 1) ? Y0 : Y1;
        for (int j = 0; j < 4; ++j) {
            // H = bf16(tanh((y + c*dt*k) @ W1^T + b1));  c=0 for k1
            gemm_kernel<1, 0><<<grid_main, blk, 0, stream>>>(
                nullptr, Yin, Kb, Cc[j], W1b, b1, 512,
                Hbf, nullptr, nullptr, nullptr, 0.0f, 0, x, s);
            // k_j = H @ W2^T + b2;  Yout = (j==0 ? Yin : Yout) + w_j*dt*k_j
            gemm_kernel<0, 1><<<grid_main, blk, 0, stream>>>(
                Hbf, nullptr, nullptr, 0.0f, W2b, b2, 512,
                nullptr, Kb, Yin, Yout, Wg[j], (j == 0) ? 1 : 0, x, s);
        }
    }

    // Final state is in Y1 (after 255 steps). out = yT @ Wout^T + bout
    gemm_kernel<1, 2><<<grid_out, blk, 0, stream>>>(
        nullptr, Y1, Y1, 0.0f, Wob, bout, 128,
        nullptr, out, nullptr, nullptr, 0.0f, 0, x, 0);
}
